// MultiHeadAttention_52209622450793
// MI455X (gfx1250) — matrix operand order repaired and checked by reading, not compiled, not run on hardware
//
#include <hip/hip_runtime.h>
#include <hip/hip_bf16.h>

// ---------------------------------------------------------------------------
// MHA forward for MI455X (gfx1250, wave32, WMMA bf16 -> f32 accumulate).
// B=2, T=2048, D=512, H=8, d_k=64.
// K/V/W tiles staged into LDS with the Tensor Data Mover (TDM), double-
// buffered and tracked with TENSORcnt.
// ---------------------------------------------------------------------------

typedef __attribute__((ext_vector_type(16))) __bf16 v16bf;
typedef __attribute__((ext_vector_type(8)))  __bf16 bf16x8;
typedef __attribute__((ext_vector_type(4)))  __bf16 bf16x4;
typedef __attribute__((ext_vector_type(8)))  float  f32x8;
typedef __attribute__((ext_vector_type(4)))  float  f32x4;
typedef __attribute__((ext_vector_type(8)))  float  v8f;
typedef __attribute__((ext_vector_type(4)))  unsigned int u32x4;
typedef __attribute__((ext_vector_type(8)))  int    i32x8;
typedef __attribute__((ext_vector_type(4)))  int    i32x4;

union V16 { v16bf v; bf16x8 h[2]; };

#define WMMA_BF16(A, Bm, C) \
  __builtin_amdgcn_wmma_f32_16x16x32_bf16(false, (A), false, (Bm), (short)0, (C), false, false)

#define D_MODEL 512
#define T_SEQ   2048
#define NHEAD   8
#define DK      64

// ---------------------------------------------------------------------------
// TDM: 2-D tile load (bf16 elements) into LDS.  D# packing per CDNA5 ISA
// ch.8: group0 = {flags, lds_addr, global_addr, type=2}; group1 = dims/strides
// (data_size=1 -> 2 bytes).  Groups 2/3 (and trailing group) zero for <=2-D.
// 6-arg builtin form (this toolchain).
// ---------------------------------------------------------------------------
__device__ __forceinline__ unsigned lds_off_u32(const void* p) {
  // flat shared pointers carry the LDS byte offset in the low 32 bits
  return (unsigned)(unsigned long long)(uintptr_t)p;
}

__device__ __forceinline__ void tdm_load_2d_bf16(unsigned lds_off, const void* gptr,
                                                 unsigned tensor_d0, unsigned tensor_d1,
                                                 unsigned tile_d0, unsigned tile_d1,
                                                 unsigned stride0) {
  const unsigned long long ga = (unsigned long long)(uintptr_t)gptr;
  u32x4 g0;
  g0[0] = 1u;                                     // count=1, user descriptor
  g0[1] = lds_off;                                // lds_addr (bytes)
  g0[2] = (unsigned)ga;                           // global_addr[31:0]
  g0[3] = (unsigned)(ga >> 32) | 0x80000000u;     // global_addr[56:32] | type=2
  i32x8 g1;
  g1[0] = 0x00010000;                             // wg_mask=0, data_size=1 (2B)
  g1[1] = (int)(tensor_d0 << 16);                 // tensor_dim0[15:0]
  g1[2] = (int)((tensor_d0 >> 16) | (tensor_d1 << 16));
  g1[3] = (int)((tensor_d1 >> 16) | (tile_d0 << 16));
  g1[4] = (int)(tile_d1 & 0xffffu);               // tile_dim1 | tile_dim2=0
  g1[5] = (int)stride0;                           // tensor_dim0_stride[31:0]
  g1[6] = 0;
  g1[7] = 0;
  i32x4 z4; z4[0] = 0; z4[1] = 0; z4[2] = 0; z4[3] = 0;
  i32x8 z8;
#pragma unroll
  for (int i = 0; i < 8; ++i) z8[i] = 0;
  __builtin_amdgcn_tensor_load_to_lds(g0, g1, z4, z4, z8, 0);
}

// ---------------------------------------------------------------------------
// fp32 -> bf16 weight conversion (512x512 per launch)
// ---------------------------------------------------------------------------
__global__ __launch_bounds__(256) void cvt_w(const float* __restrict__ src,
                                             __bf16* __restrict__ dst) {
  const int i = (blockIdx.x * 256 + threadIdx.x) * 4;
  f32x4 f = *(const f32x4*)(src + i);
  bf16x4 o;
  o[0] = (__bf16)f[0]; o[1] = (__bf16)f[1];
  o[2] = (__bf16)f[2]; o[3] = (__bf16)f[3];
  *(bf16x4*)(dst + i) = o;
}

// ---------------------------------------------------------------------------
// GEMM: out = X @ W^T + bias.  X:[4096,512], W:[512,512] row-major bf16.
// The 64x512 W block used by this WG is staged once into LDS via TDM.
// MODE 0: q proj -> bf16 [B,H,T,64] * 1/8   MODE 1: k proj -> bf16 [B,H,T,64]
// MODE 2: v proj -> bf16 [B,H,64,T]         MODE 3: o proj -> f32 [B,T,512]
// ---------------------------------------------------------------------------
template <int MODE>
__global__ __launch_bounds__(256) void proj_gemm(
    const float*  __restrict__ Xf,
    const __bf16* __restrict__ Xb,
    const __bf16* __restrict__ W,
    const float*  __restrict__ bias,
    __bf16*       __restrict__ outb,
    float*        __restrict__ outf) {
  __shared__ __bf16 wls[64 * D_MODEL];  // 64 KB: this WG's W rows
  const int lane = threadIdx.x & 31;
  const int wave = threadIdx.x >> 5;
  const int hi   = lane >> 4;
  const int ln   = lane & 15;
  const int mbase = blockIdx.x * 128 + wave * 16;
  const int nbase = blockIdx.y * 64;
  const int arow  = mbase + ln;

  // Provably-scalar issuer check: SGPR compare -> s_cbranch, so only wave 0
  // reaches the TDM issue (TDM ignores EXEC; exec-masked if-conversion would
  // have every wave issue a duplicate DMA).
  if (__builtin_amdgcn_readfirstlane(wave) == 0) {
    tdm_load_2d_bf16(lds_off_u32(&wls[0]), W + nbase * D_MODEL,
                     /*tensor*/ D_MODEL, D_MODEL, /*tile*/ D_MODEL, 64,
                     /*stride0*/ D_MODEL);
    __builtin_amdgcn_s_wait_tensorcnt(0);
  }
  __syncthreads();

  v8f acc[4] = {};

  for (int k = 0; k < D_MODEL; k += 32) {
    // A fragment: row = arow; K = k + 8*hi + {0..7} and +16+{0..7}
    V16 a;
    const int ka = k + hi * 8;
    if (MODE < 3) {
      const float* ap = Xf + arow * D_MODEL + ka;
      f32x8 f0 = *(const f32x8*)(ap);
      f32x8 f1 = *(const f32x8*)(ap + 16);
      a.h[0] = __builtin_convertvector(f0, bf16x8);
      a.h[1] = __builtin_convertvector(f1, bf16x8);
    } else {
      const __bf16* ap = Xb + arow * D_MODEL + ka;
      a.h[0] = *(const bf16x8*)(ap);
      a.h[1] = *(const bf16x8*)(ap + 16);
    }
    // B fragments from LDS-resident W rows: local row nt*16+ln, K = k+8*hi+{0..7} and +16+{0..7}
    const int kb = k + hi * 8;
    V16 bfr[4];
#pragma unroll
    for (int nt = 0; nt < 4; ++nt) {
      const __bf16* wp = &wls[(nt * 16 + ln) * D_MODEL + kb];
      bfr[nt].h[0] = *(const bf16x8*)(wp);
      bfr[nt].h[1] = *(const bf16x8*)(wp + 16);
    }
#pragma unroll
    for (int nt = 0; nt < 4; ++nt)
      acc[nt] = WMMA_BF16(a.v, bfr[nt].v, acc[nt]);
  }

  // Epilogue. C/D layout: lane holds out[m = r + 8*hi][n = ln] in acc[nt][r].
#pragma unroll
  for (int nt = 0; nt < 4; ++nt) {
    const int n = nbase + nt * 16 + ln;
    const float bv = bias[n];
#pragma unroll
    for (int r = 0; r < 8; ++r) {
      const int m = mbase + r + 8 * hi;
      float val = acc[nt][r] + bv;
      if (MODE == 0) val *= 0.125f;  // fold 1/sqrt(d_k) into q
      if (MODE == 3) {
        outf[m * D_MODEL + n] = val;
      } else {
        const int bb = m >> 11, t = m & (T_SEQ - 1);
        const int hh = n >> 6,  d = n & (DK - 1);
        if (MODE == 2)
          outb[((bb * NHEAD + hh) * DK + d) * T_SEQ + t] = (__bf16)val;   // vT
        else
          outb[((bb * NHEAD + hh) * T_SEQ + t) * DK + d] = (__bf16)val;   // q/k
      }
    }
  }
}

// ---------------------------------------------------------------------------
// Flash attention: one WG per (b*h, 128-query block); each wave owns 16 rows.
// K/V staged per 64-key chunk into LDS by TDM (double buffered); online
// softmax every 64 keys; 16 WMMA per chunk.
// ---------------------------------------------------------------------------
__global__ __launch_bounds__(256) void attn_fwd(
    const __bf16* __restrict__ qbuf,  // [B,H,T,64]
    const __bf16* __restrict__ kbuf,  // [B,H,T,64]
    const __bf16* __restrict__ vT,    // [B,H,64,T]
    __bf16*       __restrict__ obuf)  // [B,T,512]
{
  __shared__ __bf16 kls[2][64 * DK];    // 2 x 8 KB key tiles   [64 keys][64 d]
  __shared__ __bf16 vls[2][DK * 64];    // 2 x 8 KB value tiles [64 d][64 keys]
  __shared__ __bf16 plds[8][16 * 64];   // per-wave 16x64 P tile (row-major)

  const int lane = threadIdx.x & 31;
  const int wave = threadIdx.x >> 5;
  const int hi   = lane >> 4;
  const int ln   = lane & 15;
  const int bh   = blockIdx.y;
  const int bI   = bh >> 3, hI = bh & 7;
  const int qbase = blockIdx.x * 128 + wave * 16;
  const int wave_sgpr = __builtin_amdgcn_readfirstlane(wave);  // scalar wave id

  const __bf16* kb0 = kbuf + bh * T_SEQ * DK;
  const __bf16* vb0 = vT + bh * DK * T_SEQ;

  // Prologue: stage first K/V chunk (wave 0 only; scalar branch).
  if (wave_sgpr == 0) {
    tdm_load_2d_bf16(lds_off_u32(&kls[0][0]), kb0, DK, T_SEQ, DK, 64, DK);
    tdm_load_2d_bf16(lds_off_u32(&vls[0][0]), vb0, T_SEQ, DK, 64, DK, T_SEQ);
  }

  // q A-fragments for d=[0,32) and d=[32,64), resident for the whole loop.
  V16 a0, a1;
  {
    const __bf16* qrp = qbuf + (bh * T_SEQ + qbase + ln) * DK;
    const int ka = hi * 8;
    a0.h[0] = *(const bf16x8*)(qrp + ka);
    a0.h[1] = *(const bf16x8*)(qrp + ka + 16);
    a1.h[0] = *(const bf16x8*)(qrp + 32 + ka);
    a1.h[1] = *(const bf16x8*)(qrp + 32 + ka + 16);
  }

  float mrow[8], lrow[8];
#pragma unroll
  for (int r = 0; r < 8; ++r) { mrow[r] = -3.0e38f; lrow[r] = 0.0f; }
  v8f oacc[4] = {};
  __bf16* pl = &plds[wave][0];

  const int NIT = T_SEQ / 64;  // 32 chunks
  for (int it = 0; it < NIT; ++it) {
    const int j0  = it * 64;
    const int cur = it & 1;

    __syncthreads();  // everyone done reading buf[cur^1] (previous chunk)
    if (wave_sgpr == 0) {
      if (it + 1 < NIT) {
        tdm_load_2d_bf16(lds_off_u32(&kls[cur ^ 1][0]), kb0 + (j0 + 64) * DK,
                         DK, T_SEQ, DK, 64, DK);
        tdm_load_2d_bf16(lds_off_u32(&vls[cur ^ 1][0]), vb0 + (j0 + 64),
                         T_SEQ, DK, 64, DK, T_SEQ);
        __builtin_amdgcn_s_wait_tensorcnt(2);  // current pair done; next in flight
      } else {
        __builtin_amdgcn_s_wait_tensorcnt(0);
      }
    }
    __syncthreads();  // buf[cur] visible to all waves

    const __bf16* kt  = &kls[cur][0];
    const __bf16* vt_ = &vls[cur][0];

    // ---- scores: 4 key tiles x 2 d-chunks = 8 WMMA ----
    v8f s[4];
    const v8f z = {};
#pragma unroll
    for (int jt = 0; jt < 4; ++jt) {
      V16 b0, b1;
      const __bf16* kr = kt + (jt * 16 + ln) * DK + hi * 8;
      b0.h[0] = *(const bf16x8*)(kr);      b0.h[1] = *(const bf16x8*)(kr + 16);
      b1.h[0] = *(const bf16x8*)(kr + 32); b1.h[1] = *(const bf16x8*)(kr + 48);
      s[jt] = WMMA_BF16(a0.v, b0.v, z);
      s[jt] = WMMA_BF16(a1.v, b1.v, s[jt]);
    }

    // ---- online softmax over 64 keys ----
    float tmax[8], alpha[8], psum[8];
#pragma unroll
    for (int r = 0; r < 8; ++r)
      tmax[r] = fmaxf(fmaxf(s[0][r], s[1][r]), fmaxf(s[2][r], s[3][r]));
#pragma unroll
    for (int msk = 1; msk < 16; msk <<= 1)
#pragma unroll
      for (int r = 0; r < 8; ++r)
        tmax[r] = fmaxf(tmax[r], __shfl_xor(tmax[r], msk, 32));
#pragma unroll
    for (int r = 0; r < 8; ++r) {
      const float mn = fmaxf(mrow[r], tmax[r]);
      alpha[r] = __expf(mrow[r] - mn);
      mrow[r]  = mn;
      float ps = 0.0f;
#pragma unroll
      for (int jt = 0; jt < 4; ++jt) {
        const float p = __expf(s[jt][r] - mn);
        s[jt][r] = p;
        ps += p;
      }
      psum[r] = ps;
    }
#pragma unroll
    for (int msk = 1; msk < 16; msk <<= 1)
#pragma unroll
      for (int r = 0; r < 8; ++r)
        psum[r] += __shfl_xor(psum[r], msk, 32);
#pragma unroll
    for (int r = 0; r < 8; ++r) lrow[r] = lrow[r] * alpha[r] + psum[r];
#pragma unroll
    for (int t = 0; t < 4; ++t)
#pragma unroll
      for (int r = 0; r < 8; ++r) oacc[t][r] *= alpha[r];

    // ---- C-layout -> A-layout for P via wave-private LDS tile ----
#pragma unroll
    for (int jt = 0; jt < 4; ++jt)
#pragma unroll
      for (int r = 0; r < 8; ++r) {
        const int row = r + 8 * hi;
        pl[row * 64 + jt * 16 + ln] = (__bf16)s[jt][r];
      }
    asm volatile("s_wait_dscnt 0" ::: "memory");  // wave-private; DS is in-order
    V16 ap0, ap1;
    const __bf16* pr = pl + ln * 64 + hi * 8;
    ap0.h[0] = *(const bf16x8*)(pr);       ap0.h[1] = *(const bf16x8*)(pr + 16);
    ap1.h[0] = *(const bf16x8*)(pr + 32);  ap1.h[1] = *(const bf16x8*)(pr + 48);

    // ---- O += P @ V : 2 key-chunks x 4 d-tiles = 8 WMMA ----
#pragma unroll
    for (int t = 0; t < 4; ++t) {
      V16 bv0, bv1;
      const __bf16* vp = vt_ + (t * 16 + ln) * 64 + hi * 8;
      bv0.h[0] = *(const bf16x8*)(vp);      bv0.h[1] = *(const bf16x8*)(vp + 16);
      bv1.h[0] = *(const bf16x8*)(vp + 32); bv1.h[1] = *(const bf16x8*)(vp + 48);
      oacc[t] = WMMA_BF16(ap0.v, bv0.v, oacc[t]);
      oacc[t] = WMMA_BF16(ap1.v, bv1.v, oacc[t]);
    }
  }

  // ---- finalize: O /= l, write bf16 [B,T,512] for the output projection ----
#pragma unroll
  for (int r = 0; r < 8; ++r) {
    const float inv = 1.0f / lrow[r];
    const int trow = qbase + r + 8 * hi;
#pragma unroll
    for (int t = 0; t < 4; ++t) {
      const float val = oacc[t][r] * inv;
      obuf[(bI * T_SEQ + trow) * D_MODEL + hI * 64 + t * 16 + ln] = (__bf16)val;
    }
  }
}

// ---------------------------------------------------------------------------
// Launcher.  Workspace (bf16 elems): 4x262144 weights, then q/k/vT/o buffers
// (4 x 2097152).  ~19 MB total.
// ---------------------------------------------------------------------------
extern "C" void kernel_launch(void* const* d_in, const int* in_sizes, int n_in,
                              void* d_out, int out_size, void* d_ws, size_t ws_size,
                              hipStream_t stream) {
  const float* Q  = (const float*)d_in[0];
  const float* K  = (const float*)d_in[1];
  const float* V  = (const float*)d_in[2];
  const float* Wq = (const float*)d_in[3];
  const float* bq = (const float*)d_in[4];
  const float* Wk = (const float*)d_in[5];
  const float* bk = (const float*)d_in[6];
  const float* Wv = (const float*)d_in[7];
  const float* bv = (const float*)d_in[8];
  const float* Wo = (const float*)d_in[9];
  const float* bo = (const float*)d_in[10];
  float* out = (float*)d_out;

  __bf16* ws  = (__bf16*)d_ws;
  __bf16* wqb = ws;
  __bf16* wkb = ws + 1 * 262144;
  __bf16* wvb = ws + 2 * 262144;
  __bf16* wob = ws + 3 * 262144;
  __bf16* qb  = ws + 4 * 262144;
  __bf16* kb  = qb + 2097152;
  __bf16* vt  = kb + 2097152;
  __bf16* ob  = vt + 2097152;

  cvt_w<<<256, 256, 0, stream>>>(Wq, wqb);
  cvt_w<<<256, 256, 0, stream>>>(Wk, wkb);
  cvt_w<<<256, 256, 0, stream>>>(Wv, wvb);
  cvt_w<<<256, 256, 0, stream>>>(Wo, wob);

  dim3 gproj(32, 8), blk(256);
  proj_gemm<0><<<gproj, blk, 0, stream>>>(Q, nullptr, wqb, bq, qb, nullptr);
  proj_gemm<1><<<gproj, blk, 0, stream>>>(K, nullptr, wkb, bk, kb, nullptr);
  proj_gemm<2><<<gproj, blk, 0, stream>>>(V, nullptr, wvb, bv, vt, nullptr);

  attn_fwd<<<dim3(16, 16), blk, 0, stream>>>(qb, kb, vt, ob);

  proj_gemm<3><<<gproj, blk, 0, stream>>>(nullptr, ob, wob, bo, nullptr, out);
}
